// CMPNN_48404281426494
// MI455X (gfx1250) — matrix operand order repaired and checked by reading, not compiled, not run on hardware
//
#include <hip/hip_runtime.h>
#include <stdint.h>

// ---------------- problem constants (from reference) ----------------
#define HV   300      // hidden size (valid)
#define HS   320      // hidden size padded (16/32-multiple)
#define G3   960      // 3 gates * HS
#define H2   640      // 2 * HS
#define AN   40001    // atoms (incl pad row 0)
#define AP   40032    // atoms padded to 32
#define BNN  160001   // bonds
#define BP   160032   // bonds padded to 32
#define NBR  6        // neighbors
#define NM   1000     // molecules
#define TT   40       // atoms per molecule
#define MPG  1024     // NM padded to 32
#define SEQ  40000    // NM*TT (divisible by 32)
#define KF   160      // padded feature dim (133/147 -> 160)
#define KB_F 5        // 160/32
#define KB_H 10       // 320/32
#define KB_3H 30      // 960/32
#define KB_2H 20      // 640/32

typedef __attribute__((ext_vector_type(16))) __bf16         v16bf;
typedef __attribute__((ext_vector_type(8)))  float          v8f;
typedef __attribute__((ext_vector_type(8)))  unsigned short ushort8;

__device__ __forceinline__ unsigned short f2bf(float f) {
  union { float f; unsigned int u; } c; c.f = f;
  unsigned int u = c.u;
  u += 0x7FFFu + ((u >> 16) & 1u);     // round-to-nearest-even
  return (unsigned short)(u >> 16);
}

union FragU { ushort8 h[2]; v16bf v; };

__device__ __forceinline__ v16bf load_frag(const unsigned short* p0,
                                           const unsigned short* p1) {
  FragU u;
  u.h[0] = *reinterpret_cast<const ushort8*>(p0);
  u.h[1] = *reinterpret_cast<const ushort8*>(p1);
  return u.v;
}

// ---------------- WMMA GEMM ----------------
// C[Mp,Np] = act( A[Mp,lda](bf16,row-major) @ Bprep + bias + resid )
// Bprep layout per (n, kb): 32 contiguous bf16 = {half0: K=kb*32+0..15, half1: K=kb*32+16..31}
// One wave computes a 32x64 strip: 2 M-subtiles x 4 N-tiles, 8 WMMAs per K-step
// sharing 4 B fragments (22 FLOP/B from L2).
__global__ __launch_bounds__(256) void k_gemm(
    const unsigned short* __restrict__ A, const unsigned short* __restrict__ B,
    const float* __restrict__ bias, const float* __restrict__ resid,
    float* __restrict__ C, int Mp, int Np, int KBn, int lda, int relu)
{
  const int NT4 = Np >> 6;
  const int MT  = Mp >> 5;              // 32-row strips
  int gw = (int)(((size_t)blockIdx.x * blockDim.x + threadIdx.x) >> 5);
  if (gw >= MT * NT4) return;           // wave-uniform: EXEC all-1s inside
  int tm   = gw / NT4;
  int tn   = gw - tm * NT4;
  int lane = threadIdx.x & 31;
  int half = lane >> 4;
  int l    = lane & 15;

  v8f acc[2][4] = {};
  // A: lanes 0-15 need K{0..15}, lanes 16-31 K{16..31} (matches B prep layout)
  const unsigned short* arow0 = A + (size_t)(tm * 32 + l) * lda + half * 16;
  const unsigned short* arow1 = arow0 + (size_t)16 * lda;
  const unsigned short* bcol  = B + (size_t)(tn * 64 + l) * KBn * 32 + half * 16;
  const size_t nstep = (size_t)16 * KBn * 32;   // 16 columns forward in prep layout

  for (int kb = 0; kb < KBn; ++kb) {
    v16bf a0 = load_frag(arow0 + kb * 32, arow0 + kb * 32 + 8);
    v16bf a1 = load_frag(arow1 + kb * 32, arow1 + kb * 32 + 8);
    const unsigned short* bb = bcol + (size_t)kb * 32;
#pragma unroll
    for (int j = 0; j < 4; ++j) {
      v16bf bmat = load_frag(bb + (size_t)j * nstep, bb + (size_t)j * nstep + 8);
      acc[0][j] = __builtin_amdgcn_wmma_f32_16x16x32_bf16(
          false, a0, false, bmat, (short)0, acc[0][j], false, false);
      acc[1][j] = __builtin_amdgcn_wmma_f32_16x16x32_bf16(
          false, a1, false, bmat, (short)0, acc[1][j], false, false);
    }
  }

#pragma unroll
  for (int s = 0; s < 2; ++s) {
    int m0 = tm * 32 + s * 16 + half * 8;   // C/D: vgpr r -> M = r + half*8
#pragma unroll
    for (int j = 0; j < 4; ++j) {
      int n = tn * 64 + j * 16 + l;
      float bv = bias ? bias[n] : 0.0f;
#pragma unroll
      for (int r = 0; r < 8; ++r) {
        size_t idx = (size_t)(m0 + r) * Np + n;
        float v = acc[s][j][r] + bv;
        if (resid) v += resid[idx];
        if (relu)  v = v > 0.0f ? v : 0.0f;
        C[idx] = v;
      }
    }
  }
}

// ---------------- weight / bias preparation ----------------
// direct (gated along K): B[k][n] = W[(g*Kv + c)*Nv + n], k = g*KgPad + c
__global__ void k_prep_w(const float* __restrict__ W, unsigned short* __restrict__ dst,
                         int KBn, int Np, int KgPad, int Kv, int Kg, int Nv) {
  size_t i = (size_t)blockIdx.x * blockDim.x + threadIdx.x;
  size_t total = (size_t)Np * KBn * 32;
  if (i >= total) return;
  int e    = (int)(i & 15);
  int half = (int)((i >> 4) & 1);
  size_t rest = i >> 5;
  int kb = (int)(rest % KBn);
  int n  = (int)(rest / KBn);
  int k  = kb * 32 + half * 16 + e;     // B 32x16 as k_gemm loads it: lane=col, K=half*16+e
  int g = k / KgPad, c = k % KgPad;
  float v = 0.0f;
  if (g < Kg && c < Kv && n < Nv) v = W[(size_t)(g * Kv + c) * Nv + n];
  dst[i] = f2bf(v);
}

// transposed GRU weights: B[k][n] = W[(g*HV + c)*HV + k]  (gates along N, Np=960)
__global__ void k_prep_wT(const float* __restrict__ W, unsigned short* __restrict__ dst) {
  size_t i = (size_t)blockIdx.x * blockDim.x + threadIdx.x;
  size_t total = (size_t)G3 * KB_H * 32;
  if (i >= total) return;
  int e    = (int)(i & 15);
  int half = (int)((i >> 4) & 1);
  size_t rest = i >> 5;
  int kb = (int)(rest % KB_H);
  int n  = (int)(rest / KB_H);
  int k  = kb * 32 + half * 16 + e;
  int g = n / HS, c = n % HS;
  float v = 0.0f;
  if (k < HV && c < HV) v = W[(size_t)(g * HV + c) * HV + k];
  dst[i] = f2bf(v);
}

__global__ void k_prep_bias(const float* __restrict__ b, float* __restrict__ dst, int Np) {
  int i = blockIdx.x * blockDim.x + threadIdx.x;
  if (i >= Np) return;
  int g = i / HS, c = i % HS;
  dst[i] = (c < HV) ? b[g * HV + c] : 0.0f;
}

// f32 [Mv,Kv] -> zero-padded bf16 [Mp,Kp]
__global__ void k_pad_bf16(const float* __restrict__ src, unsigned short* __restrict__ dst,
                           int Mv, int Kv, int Mp, int Kp) {
  size_t i = (size_t)blockIdx.x * blockDim.x + threadIdx.x;
  size_t total = (size_t)Mp * Kp;
  if (i >= total) return;
  int k = (int)(i % Kp);
  size_t m = i / Kp;
  float v = (m < (size_t)Mv && k < Kv) ? src[m * Kv + k] : 0.0f;
  dst[i] = f2bf(v);
}

// ---------------- message passing elementwise ----------------
__global__ void k_agg_accum(const float* __restrict__ mbond, const int* __restrict__ a2b,
                            float* __restrict__ matom) {
  size_t i = (size_t)blockIdx.x * blockDim.x + threadIdx.x;
  if (i >= (size_t)AP * HS) return;
  int h = (int)(i % HS);
  size_t a = i / HS;
  if (a >= AN) return;                     // pad rows stay 0
  float s = 0.0f, mx = -3.0e38f;
#pragma unroll
  for (int j = 0; j < NBR; ++j) {
    int b = a2b[a * NBR + j];
    float v = mbond[(size_t)b * HS + h];
    s += v; mx = fmaxf(mx, v);
  }
  matom[i] += s * mx;
}

__global__ void k_agg_concat(const float* __restrict__ mbond, const int* __restrict__ a2b,
                             const float* __restrict__ matom, const float* __restrict__ iatom,
                             unsigned short* __restrict__ agg3) {
  size_t i = (size_t)blockIdx.x * blockDim.x + threadIdx.x;
  if (i >= (size_t)AP * HS) return;
  int h = (int)(i % HS);
  size_t a = i / HS;
  float agg = 0.0f, ma = 0.0f, ia = 0.0f;
  if (a < AN) {
    float s = 0.0f, mx = -3.0e38f;
#pragma unroll
    for (int j = 0; j < NBR; ++j) {
      int b = a2b[a * NBR + j];
      float v = mbond[(size_t)b * HS + h];
      s += v; mx = fmaxf(mx, v);
    }
    agg = s * mx;
    ma = matom[i];
    ia = iatom[i];
  }
  agg3[a * G3 + h]          = f2bf(agg);
  agg3[a * G3 + HS + h]     = f2bf(ma);
  agg3[a * G3 + 2 * HS + h] = f2bf(ia);
}

__global__ void k_mb(const float* __restrict__ matom, const float* __restrict__ mbond,
                     const int* __restrict__ b2a, const int* __restrict__ b2revb,
                     unsigned short* __restrict__ mb) {
  size_t i = (size_t)blockIdx.x * blockDim.x + threadIdx.x;
  if (i >= (size_t)BP * HS) return;
  int h = (int)(i % HS);
  size_t b = i / HS;
  float v = 0.0f;
  if (b < BNN)
    v = matom[(size_t)b2a[b] * HS + h] - mbond[(size_t)b2revb[b] * HS + h];
  mb[i] = f2bf(v);
}

// ---------------- GRU support ----------------
__global__ void k_msg_seq(const float* __restrict__ aggW, const float* __restrict__ gru_bias,
                          float* __restrict__ msg0, unsigned short* __restrict__ seq) {
  size_t i = (size_t)blockIdx.x * blockDim.x + threadIdx.x;
  if (i >= (size_t)(SEQ + 1) * HS) return;
  int h = (int)(i % HS);
  size_t r = i / HS;
  float bias = (h < HV) ? gru_bias[h] : 0.0f;
  float v = fmaxf(aggW[r * HS + h] + bias, 0.0f);
  if (r == 0) msg0[h] = v;
  else        seq[(r - 1) * HS + h] = f2bf(v);
}

__global__ void k_h0(const float* __restrict__ aggW, float* __restrict__ h0) {
  int i = blockIdx.x * blockDim.x + threadIdx.x;
  if (i >= NM * HS) return;
  int h = i % HS, mol = i / HS;
  const float* base = aggW + (size_t)(1 + mol * TT) * HS + h;
  float mx = -3.0e38f;
#pragma unroll 4
  for (int t = 0; t < TT; ++t) mx = fmaxf(mx, base[(size_t)t * HS]);
  h0[i] = mx;
}

__global__ void k_h_init(const float* __restrict__ h0, float* __restrict__ h,
                         unsigned short* __restrict__ hbf) {
  int i = blockIdx.x * blockDim.x + threadIdx.x;
  if (i >= MPG * HS) return;
  float v = (i < NM * HS) ? h0[i] : 0.0f;
  h[i] = v;
  hbf[i] = f2bf(v);
}

__global__ void k_gru_step(const float* __restrict__ xp, const float* __restrict__ gh,
                           float* __restrict__ h, unsigned short* __restrict__ hbf,
                           float* __restrict__ out, int u) {
  int i = blockIdx.x * blockDim.x + threadIdx.x;
  if (i >= NM * HS) return;
  int hh = i % HS, mol = i / HS;
  size_t row = (size_t)mol * TT + u;
  const float* xr = xp + row * G3;
  const float* gr = gh + (size_t)mol * G3;
  float r = 1.0f / (1.0f + __expf(-(xr[hh] + gr[hh])));
  float z = 1.0f / (1.0f + __expf(-(xr[HS + hh] + gr[HS + hh])));
  float n = tanhf(xr[2 * HS + hh] + r * gr[2 * HS + hh]);
  float hp = h[i];
  float hn = (1.0f - z) * n + z * hp;
  h[i] = hn;
  hbf[i] = f2bf(hn);
  out[row * HS + hh] = hn;
}

__global__ void k_msg2(const float* __restrict__ msg0, const float* __restrict__ outf,
                       const float* __restrict__ outb, unsigned short* __restrict__ m2) {
  size_t i = (size_t)blockIdx.x * blockDim.x + threadIdx.x;
  if (i >= (size_t)AP * H2) return;
  int c = (int)(i % H2);
  size_t r = i / H2;
  float v = 0.0f;
  if (r == 0)            v = msg0[c % HS];
  else if (r <= SEQ)     v = (c < HS) ? outf[(r - 1) * HS + c]
                                      : outb[(r - 1) * HS + (c - HS)];
  m2[i] = f2bf(v);
}

__global__ void k_molmean(const float* __restrict__ ah, float* __restrict__ out) {
  int i = blockIdx.x * blockDim.x + threadIdx.x;
  if (i >= NM * HV) return;
  int h = i % HV, mol = i / HV;
  const float* base = ah + (size_t)(1 + mol * TT) * HS + h;
  float s = 0.0f;
#pragma unroll 4
  for (int t = 0; t < TT; ++t) s += base[(size_t)t * HS];
  out[mol * HV + h] = s * (1.0f / TT);
}

// ---------------- host orchestration ----------------
#define L1D(kern, total, ...) do {                                          \
    size_t _t = (size_t)(total);                                            \
    unsigned _b = (unsigned)((_t + 255) / 256);                             \
    kern<<<dim3(_b), dim3(256), 0, stream>>>(__VA_ARGS__);                  \
  } while (0)

extern "C" void kernel_launch(void* const* d_in, const int* in_sizes, int n_in,
                              void* d_out, int out_size, void* d_ws, size_t ws_size,
                              hipStream_t stream) {
  (void)in_sizes; (void)n_in; (void)out_size; (void)ws_size;
  const float* f_atoms  = (const float*)d_in[0];
  const float* f_bonds  = (const float*)d_in[1];
  const float* Wi_atom  = (const float*)d_in[2];
  const float* Wi_bond  = (const float*)d_in[3];
  const float* Wh[2]    = { (const float*)d_in[4], (const float*)d_in[5] };
  const float* W_lr     = (const float*)d_in[6];
  const float* W_o      = (const float*)d_in[7];
  const float* b_o      = (const float*)d_in[8];
  const float* gru_bias = (const float*)d_in[9];
  const float* gWih_f   = (const float*)d_in[10];
  const float* gWhh_f   = (const float*)d_in[11];
  const float* gbih_f   = (const float*)d_in[12];
  const float* gbhh_f   = (const float*)d_in[13];
  const float* gWih_b   = (const float*)d_in[14];
  const float* gWhh_b   = (const float*)d_in[15];
  const float* gbih_b   = (const float*)d_in[16];
  const float* gbhh_b   = (const float*)d_in[17];
  const int*   a2b      = (const int*)d_in[18];
  const int*   b2a      = (const int*)d_in[19];
  const int*   b2revb   = (const int*)d_in[20];

  // workspace arena (recomputed identically every call)
  char* base = (char*)d_ws;
  size_t off = 0;
  auto take = [&](size_t bytes) -> void* {
    off = (off + 255) & ~(size_t)255;
    void* p = base + off; off += bytes; return p;
  };
  unsigned short* fa_bf = (unsigned short*)take((size_t)AP * KF * 2);
  unsigned short* fb_bf = (unsigned short*)take((size_t)BP * KF * 2);
  unsigned short* wiA   = (unsigned short*)take((size_t)HS * KB_F * 32 * 2);
  unsigned short* wiB   = (unsigned short*)take((size_t)HS * KB_F * 32 * 2);
  unsigned short* whp0  = (unsigned short*)take((size_t)HS * KB_H * 32 * 2);
  unsigned short* whp1  = (unsigned short*)take((size_t)HS * KB_H * 32 * 2);
  unsigned short* wlrp  = (unsigned short*)take((size_t)HS * KB_3H * 32 * 2);
  unsigned short* wop   = (unsigned short*)take((size_t)HS * KB_2H * 32 * 2);
  unsigned short* wihF  = (unsigned short*)take((size_t)G3 * KB_H * 32 * 2);
  unsigned short* whhF  = (unsigned short*)take((size_t)G3 * KB_H * 32 * 2);
  unsigned short* wihB  = (unsigned short*)take((size_t)G3 * KB_H * 32 * 2);
  unsigned short* whhB  = (unsigned short*)take((size_t)G3 * KB_H * 32 * 2);
  float* bbo   = (float*)take((size_t)HS * 4);
  float* bihF  = (float*)take((size_t)G3 * 4);
  float* bhhF  = (float*)take((size_t)G3 * 4);
  float* bihB  = (float*)take((size_t)G3 * 4);
  float* bhhB  = (float*)take((size_t)G3 * 4);
  float* iatom = (float*)take((size_t)AP * HS * 4);
  float* matom = (float*)take((size_t)AP * HS * 4);
  float* ibond = (float*)take((size_t)BP * HS * 4);
  float* mbond = (float*)take((size_t)BP * HS * 4);
  unsigned short* mbbf = (unsigned short*)take((size_t)BP * HS * 2);
  unsigned short* agg3 = (unsigned short*)take((size_t)AP * G3 * 2);
  float* aggW  = (float*)take((size_t)AP * HS * 4);
  unsigned short* seqbf = (unsigned short*)take((size_t)SEQ * HS * 2);
  float* msg0  = (float*)take((size_t)HS * 4);
  float* h0    = (float*)take((size_t)NM * HS * 4);
  float* hbuf  = (float*)take((size_t)MPG * HS * 4);
  unsigned short* hbf = (unsigned short*)take((size_t)MPG * HS * 2);
  float* gh    = (float*)take((size_t)MPG * G3 * 4);
  float* xpf   = (float*)take((size_t)SEQ * G3 * 4);
  float* xpb   = (float*)take((size_t)SEQ * G3 * 4);
  float* outf  = (float*)take((size_t)SEQ * HS * 4);
  float* outb  = (float*)take((size_t)SEQ * HS * 4);
  unsigned short* m2bf = (unsigned short*)take((size_t)AP * H2 * 2);
  float* ahid  = (float*)take((size_t)AP * HS * 4);

  auto gemm = [&](const unsigned short* A, const unsigned short* B, const float* bias,
                  const float* resid, float* C, int Mp, int Np, int KBn, int lda, int relu) {
    int waves = (Mp / 32) * (Np / 64);
    k_gemm<<<dim3((unsigned)((waves + 7) / 8)), dim3(256), 0, stream>>>(
        A, B, bias, resid, C, Mp, Np, KBn, lda, relu);
  };

  // --- weight / input preparation ---
  L1D(k_pad_bf16, (size_t)AP * KF, f_atoms, fa_bf, AN, 133, AP, KF);
  L1D(k_pad_bf16, (size_t)BP * KF, f_bonds, fb_bf, BNN, 147, BP, KF);
  L1D(k_prep_w, (size_t)HS * KB_F * 32, Wi_atom, wiA, KB_F, HS, KF, 133, 1, HV);
  L1D(k_prep_w, (size_t)HS * KB_F * 32, Wi_bond, wiB, KB_F, HS, KF, 147, 1, HV);
  L1D(k_prep_w, (size_t)HS * KB_H * 32, Wh[0], whp0, KB_H, HS, HS, HV, 1, HV);
  L1D(k_prep_w, (size_t)HS * KB_H * 32, Wh[1], whp1, KB_H, HS, HS, HV, 1, HV);
  L1D(k_prep_w, (size_t)HS * KB_3H * 32, W_lr, wlrp, KB_3H, HS, HS, HV, 3, HV);
  L1D(k_prep_w, (size_t)HS * KB_2H * 32, W_o,  wop,  KB_2H, HS, HS, HV, 2, HV);
  L1D(k_prep_wT, (size_t)G3 * KB_H * 32, gWih_f, wihF);
  L1D(k_prep_wT, (size_t)G3 * KB_H * 32, gWhh_f, whhF);
  L1D(k_prep_wT, (size_t)G3 * KB_H * 32, gWih_b, wihB);
  L1D(k_prep_wT, (size_t)G3 * KB_H * 32, gWhh_b, whhB);
  L1D(k_prep_bias, HS, b_o, bbo, HS);
  L1D(k_prep_bias, G3, gbih_f, bihF, G3);
  L1D(k_prep_bias, G3, gbhh_f, bhhF, G3);
  L1D(k_prep_bias, G3, gbih_b, bihB, G3);
  L1D(k_prep_bias, G3, gbhh_b, bhhB, G3);

  // --- input projections ---
  gemm(fa_bf, wiA, nullptr, nullptr, iatom, AP, HS, KB_F, KF, 1);   // relu(f_atoms@Wi_atom)
  gemm(fb_bf, wiB, nullptr, nullptr, ibond, BP, HS, KB_F, KF, 1);   // relu(f_bonds@Wi_bond)
  hipMemcpyAsync(matom, iatom, (size_t)AP * HS * 4, hipMemcpyDeviceToDevice, stream);
  hipMemcpyAsync(mbond, ibond, (size_t)BP * HS * 4, hipMemcpyDeviceToDevice, stream);

  // --- 2 message passing steps ---
  const unsigned short* whp[2] = { whp0, whp1 };
  for (int it = 0; it < 2; ++it) {
    L1D(k_agg_accum, (size_t)AP * HS, mbond, a2b, matom);
    L1D(k_mb, (size_t)BP * HS, matom, mbond, b2a, b2revb, mbbf);
    gemm(mbbf, whp[it], nullptr, ibond, mbond, BP, HS, KB_H, HS, 1); // relu(ibond + mb@Wh)
  }

  // --- final aggregation + W_lr ---
  L1D(k_agg_concat, (size_t)AP * HS, mbond, a2b, matom, iatom, agg3);
  gemm(agg3, wlrp, nullptr, nullptr, aggW, AP, HS, KB_3H, G3, 0);

  // --- GRU prep ---
  L1D(k_msg_seq, (size_t)(SEQ + 1) * HS, aggW, gru_bias, msg0, seqbf);
  L1D(k_h0, NM * HS, aggW, h0);
  gemm(seqbf, wihF, bihF, nullptr, xpf, SEQ, G3, KB_H, HS, 0);      // x@Wih_f.T + bih_f
  gemm(seqbf, wihB, bihB, nullptr, xpb, SEQ, G3, KB_H, HS, 0);      // x@Wih_b.T + bih_b

  // --- forward GRU ---
  L1D(k_h_init, MPG * HS, h0, hbuf, hbf);
  for (int t = 0; t < TT; ++t) {
    gemm(hbf, whhF, bhhF, nullptr, gh, MPG, G3, KB_H, HS, 0);       // h@Whh_f.T + bhh_f
    L1D(k_gru_step, NM * HS, xpf, gh, hbuf, hbf, outf, t);
  }
  // --- backward GRU ---
  L1D(k_h_init, MPG * HS, h0, hbuf, hbf);
  for (int s = 0; s < TT; ++s) {
    gemm(hbf, whhB, bhhB, nullptr, gh, MPG, G3, KB_H, HS, 0);       // h@Whh_b.T + bhh_b
    L1D(k_gru_step, NM * HS, xpb, gh, hbuf, hbf, outb, TT - 1 - s);
  }

  // --- output head ---
  L1D(k_msg2, (size_t)AP * H2, msg0, outf, outb, m2bf);
  gemm(m2bf, wop, bbo, nullptr, ahid, AP, HS, KB_2H, H2, 1);        // relu(m2@W_o + b_o)
  L1D(k_molmean, NM * HV, ahid, (float*)d_out);
}
